// MultiScaleRetention_25254407700882
// MI455X (gfx1250) — compile-verified, matrix operand order checked by reading, not run on hardware
//
#include <hip/hip_runtime.h>
#include <hip/hip_bf16.h>
#include <math.h>
#include <stdint.h>

// ---------------- problem constants ----------------
#define B_  2
#define T_  4096
#define D_  2048
#define H_  16
#define CL_ 512
#define DK_ 128
#define NC_ 8
#define SCALING_ 0.08838834764831845f   // DK^-0.5

typedef __bf16 bf16_t;
typedef __attribute__((ext_vector_type(16))) __bf16 bf16x16;
typedef __attribute__((ext_vector_type(8)))  __bf16 bf16x8;
typedef __attribute__((ext_vector_type(8)))  float  v8f;

// ---------------- WMMA fragment helpers ----------------
// A/B 16x32 bf16 fragment from row-major bf16 source.
// lanes 0-15: rows 0-15, K 0..7 & 16..23 ; lanes 16-31: rows 0-15, K 8..15 & 24..31.
__device__ __forceinline__ bf16x16 frag_bf16(const bf16_t* base, int ld, int row0, int k0) {
    int lane = threadIdx.x & 31, half = lane >> 4, r = lane & 15;
    const bf16_t* p = base + (size_t)(row0 + r) * ld + k0 + half * 8;
    union { bf16x16 v; bf16x8 h[2]; } u;
    u.h[0] = *(const bf16x8*)p;
    u.h[1] = *(const bf16x8*)(p + 16);
    return u.v;
}

// same layout, f32 source converted on the fly, optional per-lane multiplier
__device__ __forceinline__ bf16x16 frag_f32(const float* base, int ld, int row0, int k0, float mul) {
    int lane = threadIdx.x & 31, half = lane >> 4, r = lane & 15;
    const float* p = base + (size_t)(row0 + r) * ld + k0 + half * 8;
    bf16x16 out;
#pragma unroll
    for (int i = 0; i < 8; ++i) out[i]     = (__bf16)(p[i]      * mul);
#pragma unroll
    for (int i = 0; i < 8; ++i) out[8 + i] = (__bf16)(p[16 + i] * mul);
    return out;
}

// strided-K gather fragment: element(row=lane&15 (+n0), k) = base[k*ldRow + n0 + (lane&15)]
__device__ __forceinline__ bf16x16 frag_gather(const float* base, int ldRow, int k0, int n0,
                                               const float* wk) {
    int lane = threadIdx.x & 31, half = lane >> 4, nn = lane & 15;
    bf16x16 out;
#pragma unroll
    for (int i = 0; i < 8; ++i) {
        int k = k0 + half * 8 + i;
        float v = base[(size_t)k * ldRow + n0 + nn];
        if (wk) v *= wk[k];
        out[i] = (__bf16)v;
    }
#pragma unroll
    for (int i = 0; i < 8; ++i) {
        int k = k0 + half * 8 + 16 + i;
        float v = base[(size_t)k * ldRow + n0 + nn];
        if (wk) v *= wk[k];
        out[8 + i] = (__bf16)v;
    }
    return out;
}

#define WMMA_BF16(a, b, c) \
    __builtin_amdgcn_wmma_f32_16x16x32_bf16(false, (a), false, (b), (short)0, (c), false, false)

// ---- gfx1250 async global->LDS copy (ASYNCcnt) ----
// generic shared pointer's low 32 bits are the LDS byte offset (aperture in high bits).
__device__ __forceinline__ void async_ld_b128(const void* gsrc, void* ldst) {
    unsigned l = (unsigned)(uintptr_t)ldst;
    asm volatile("global_load_async_to_lds_b128 %0, %1, off" :: "v"(l), "v"(gsrc) : "memory");
}
__device__ __forceinline__ void wait_async0() {
    asm volatile("s_wait_asynccnt 0x0" ::: "memory");
}
__device__ __forceinline__ void wait_async4() {
    asm volatile("s_wait_asynccnt 0x4" ::: "memory");
}

// ---------------- elementwise kernels ----------------
__global__ void msr_f2bf(const float* __restrict__ in, bf16_t* __restrict__ out, size_t n) {
    size_t i = (size_t)blockIdx.x * blockDim.x + threadIdx.x;
    if (i < n) out[i] = (__bf16)in[i];
}

// in-place RoPE-like theta shift on a (B,T,D) tensor; sin/cos are (T,DK)
__global__ void msr_theta_shift(float* __restrict__ x, const float* __restrict__ sin_,
                                const float* __restrict__ cos_) {
    size_t p = (size_t)blockIdx.x * blockDim.x + threadIdx.x;   // pair index
    size_t npairs = (size_t)B_ * T_ * (D_ / 2);
    if (p >= npairs) return;
    size_t row  = p / (D_ / 2);
    int    col0 = (int)(p % (D_ / 2)) * 2;
    int    t    = (int)(row % T_);
    int    d0   = col0 % DK_;
    size_t base = row * D_ + col0;
    float x0 = x[base], x1 = x[base + 1];
    float c0 = cos_[(size_t)t * DK_ + d0], c1 = cos_[(size_t)t * DK_ + d0 + 1];
    float s0 = sin_[(size_t)t * DK_ + d0], s1 = sin_[(size_t)t * DK_ + d0 + 1];
    x[base]     = x0 * c0 - x1 * s0;
    x[base + 1] = x1 * c1 + x0 * s1;
}

// ---------------- WMMA GEMM: Y = (X @ W^T + bias) * scale ----------------
// Double-buffered async-to-LDS pipeline. Block = 256 threads (8 waves),
// tile 128(M) x 128(N); wave w -> rows [w*16,+16), 8 N-subtiles; K step 32.
__device__ __forceinline__ void gemm_stage(const bf16_t* X, const bf16_t* W, int K,
                                           int m0b, int n0b, bf16_t* sA, bf16_t* sB,
                                           int k0, int tid) {
    // 512 16B chunks per matrix (128 rows x 64B); thread does chunks tid, tid+256
#pragma unroll
    for (int c = 0; c < 2; ++c) {
        int chunk = tid + c * 256;
        int row = chunk >> 2, seg = chunk & 3;
        async_ld_b128(X + (size_t)(m0b + row) * K + k0 + seg * 8, sA + row * 32 + seg * 8);
        async_ld_b128(W + (size_t)(n0b + row) * K + k0 + seg * 8, sB + row * 32 + seg * 8);
    }
}

__global__ void __launch_bounds__(256)
msr_gemm_bf16(const bf16_t* __restrict__ X, const bf16_t* __restrict__ W,
              const float* __restrict__ bias, float* __restrict__ Y,
              int M, int N, int K, float scale) {
    __shared__ bf16_t sA[2][128 * 32];   // 8KB per buffer
    __shared__ bf16_t sB[2][128 * 32];
    const int tid = threadIdx.x;
    const int wave = tid >> 5, lane = tid & 31, half = lane >> 4, col = lane & 15;
    const int m0b = blockIdx.x * 128;
    const int n0b = blockIdx.y * 128;

    v8f acc[8];
#pragma unroll
    for (int j = 0; j < 8; ++j) acc[j] = (v8f){};

    const int steps = K / 32;
    gemm_stage(X, W, K, m0b, n0b, sA[0], sB[0], 0, tid);
    for (int ks = 0; ks < steps; ++ks) {
        int cur = ks & 1;
        if (ks + 1 < steps) {
            gemm_stage(X, W, K, m0b, n0b, sA[1 - cur], sB[1 - cur], (ks + 1) * 32, tid);
            wait_async4();          // current batch (4 older async ops) complete
        } else {
            wait_async0();
        }
        __syncthreads();
        // issue ALL fragment loads before any WMMA so ds loads overlap the MFU
        bf16x16 a = frag_bf16(sA[cur], 32, wave * 16, 0);
        bf16x16 bf[8];
#pragma unroll
        for (int j = 0; j < 8; ++j) bf[j] = frag_bf16(sB[cur], 32, j * 16, 0);
#pragma unroll
        for (int j = 0; j < 8; ++j) acc[j] = WMMA_BF16(a, bf[j], acc[j]);
        __syncthreads();            // all waves done reading before buffer refill
    }
    int mrow = m0b + wave * 16;
#pragma unroll
    for (int j = 0; j < 8; ++j) {
        int nn = n0b + j * 16 + col;
        float bb = bias[nn];
#pragma unroll
        for (int i = 0; i < 8; ++i)
            Y[(size_t)(mrow + i + 8 * half) * N + nn] = (acc[j][i] + bb) * scale;
    }
}

// ---------------- per-chunk attention inner part ----------------
// grid (H, NC, B); block 256 (8 waves). qr/kr/v are (B,T,D) with head cols h*DK+d.
__global__ void __launch_bounds__(256)
msr_attn_inner(const float* __restrict__ qr, const float* __restrict__ kr,
               const float* __restrict__ v, const float* __restrict__ mask,
               float* __restrict__ inner_out, float* __restrict__ inner_scale) {
    __shared__ float stage[8][16][32];   // per-wave qk staging (16KB)
    __shared__ float vtile[32 * 128];    // shared v tile per m-step (16KB)
    int b = blockIdx.z, n = blockIdx.y, h = blockIdx.x;
    size_t rowbase = ((size_t)b * T_ + (size_t)n * CL_) * D_ + (size_t)h * DK_;
    const float* qb = qr + rowbase;
    const float* kb = kr + rowbase;
    const float* vb = v + rowbase;
    const float* mh = mask + (size_t)h * CL_ * CL_;
    int tid = threadIdx.x;
    int wave = tid >> 5;
    int lane = tid & 31, half = lane >> 4, col = lane & 15;
    float* stg = &stage[wave][0][0];

    for (int ls = 0; ls < 4; ++ls) {
        int l0 = (ls * 8 + wave) * 16;
        // ---- pass 1: inner_scale = clip(sum_m |qk*mask|, 1) per row ----
        float rs[8];
#pragma unroll
        for (int i = 0; i < 8; ++i) rs[i] = 0.f;
        for (int mt = 0; mt < CL_; mt += 16) {
            v8f c = {};
#pragma unroll
            for (int k0 = 0; k0 < DK_; k0 += 32) {
                bf16x16 a  = frag_f32(qb, D_, l0, k0, 1.f);
                bf16x16 bf = frag_f32(kb, D_, mt, k0, 1.f);
                c = WMMA_BF16(a, bf, c);
            }
#pragma unroll
            for (int i = 0; i < 8; ++i) {
                int l = l0 + i + 8 * half;
                rs[i] += fabsf(c[i] * mh[(size_t)l * CL_ + mt + col]);
            }
        }
        float inv[8];
#pragma unroll
        for (int i = 0; i < 8; ++i) {
            float s = rs[i];
            s += __shfl_xor(s, 1, 32); s += __shfl_xor(s, 2, 32);
            s += __shfl_xor(s, 4, 32); s += __shfl_xor(s, 8, 32);
            float sc = fmaxf(s, 1.f);
            inv[i] = 1.f / sc;
            if (col == 0) {
                int l = l0 + i + 8 * half;
                inner_scale[(((size_t)b * NC_ + n) * H_ + h) * CL_ + l] = sc;
            }
        }
        // ---- pass 2: inner_out = (qk*mask/scale) @ vc ----
        v8f acc[8];
#pragma unroll
        for (int j = 0; j < 8; ++j) acc[j] = (v8f){};
        for (int m0 = 0; m0 < CL_; m0 += 32) {
            // stage v rows [m0, m0+128) x 32 cols via async-to-LDS (shared by all waves)
#pragma unroll
            for (int c = 0; c < 4; ++c) {
                int chunk = tid + c * 256;              // 1024 chunks of 16B
                int row = chunk >> 3, seg = chunk & 7;  // 8 x 16B per 128-float row
                async_ld_b128(vb + (size_t)(m0 + row) * D_ + seg * 4,
                              &vtile[row * 128 + seg * 4]);
            }
            // recompute qk tiles (scaled+masked) into per-wave staging
#pragma unroll
            for (int sub = 0; sub < 2; ++sub) {
                int mt = m0 + sub * 16;
                v8f c = {};
#pragma unroll
                for (int k0 = 0; k0 < DK_; k0 += 32) {
                    bf16x16 a  = frag_f32(qb, D_, l0, k0, 1.f);
                    bf16x16 bf = frag_f32(kb, D_, mt, k0, 1.f);
                    c = WMMA_BF16(a, bf, c);
                }
#pragma unroll
                for (int i = 0; i < 8; ++i) {
                    int l = l0 + i + 8 * half;
                    stg[(i + 8 * half) * 32 + sub * 16 + col] =
                        c[i] * mh[(size_t)l * CL_ + mt + col] * inv[i];
                }
            }
            asm volatile("s_wait_dscnt 0" ::: "memory");   // in-wave cross-lane LDS RAW
            wait_async0();                                  // v tile resident
            __syncthreads();
            // batch all fragment builds before the WMMA burst
            bf16x16 aq = frag_f32(stg, 32, 0, 0, 1.f);
            bf16x16 bv2[8];
#pragma unroll
            for (int j = 0; j < 8; ++j) bv2[j] = frag_gather(vtile, 128, 0, j * 16, nullptr);
#pragma unroll
            for (int j = 0; j < 8; ++j) acc[j] = WMMA_BF16(aq, bv2[j], acc[j]);
            __syncthreads();        // protect vtile before next refill
        }
#pragma unroll
        for (int j = 0; j < 8; ++j)
#pragma unroll
            for (int i = 0; i < 8; ++i) {
                int l = l0 + i + 8 * half;
                inner_out[rowbase + (size_t)l * D_ + j * 16 + col] = acc[j][i];
            }
    }
}

// ---------------- per-chunk kv = kr^T @ (vc * mask[-1]) ----------------
__global__ void __launch_bounds__(256)
msr_chunk_kv(const float* __restrict__ kr, const float* __restrict__ v,
             const float* __restrict__ mask, float* __restrict__ kvc) {
    int b = blockIdx.z, n = blockIdx.y, h = blockIdx.x;
    size_t rowbase = ((size_t)b * T_ + (size_t)n * CL_) * D_ + (size_t)h * DK_;
    const float* kb = kr + rowbase;
    const float* vb = v + rowbase;
    const float* wl = mask + ((size_t)h * CL_ + (CL_ - 1)) * CL_;  // mask[h, CL-1, :]
    int wave = threadIdx.x >> 5;
    int lane = threadIdx.x & 31, half = lane >> 4, col = lane & 15;
    int kd0 = wave * 16;
    v8f acc[8];
#pragma unroll
    for (int j = 0; j < 8; ++j) acc[j] = (v8f){};
    for (int l0 = 0; l0 < CL_; l0 += 32) {
        bf16x16 a = frag_gather(kb, D_, l0, kd0, nullptr);       // A[kd,l] = kr[l,kd]
        bf16x16 bb[8];
#pragma unroll
        for (int j = 0; j < 8; ++j) bb[j] = frag_gather(vb, D_, l0, j * 16, wl);
#pragma unroll
        for (int j = 0; j < 8; ++j) acc[j] = WMMA_BF16(a, bb[j], acc[j]);
    }
    size_t obase = (((size_t)b * NC_ + n) * H_ + h) * (size_t)(DK_ * DK_);
#pragma unroll
    for (int j = 0; j < 8; ++j)
#pragma unroll
        for (int i = 0; i < 8; ++i)
            kvc[obase + (size_t)(kd0 + i + 8 * half) * DK_ + j * 16 + col] = acc[j][i];
}

// ---------------- sequential chunk scan (NC=8) ----------------
__global__ void msr_kv_scan(const float* __restrict__ kvc, const float* __restrict__ cross_decay,
                            float* __restrict__ kvrec, float* __restrict__ cross_scale) {
    extern __shared__ float smem[];
    float* st      = smem;
    float* colsum  = smem + DK_ * DK_;
    float* scale_s = colsum + DK_;
    int b = blockIdx.y, h = blockIdx.x;
    int tid = threadIdx.x;
    float cd = cross_decay[h];
    for (int i = tid; i < DK_ * DK_; i += 256) st[i] = 0.f;
    if (tid == 0) *scale_s = 1.f;
    __syncthreads();
    for (int n = 0; n < NC_; ++n) {
        size_t base = (((size_t)b * NC_ + n) * H_ + h) * (size_t)(DK_ * DK_);
        float sc = *scale_s;
        for (int i = tid; i < DK_ * DK_; i += 256) kvrec[base + i] = st[i] / sc;
        if (tid == 0) cross_scale[((size_t)b * NC_ + n) * H_ + h] = sc;
        for (int i = tid; i < DK_ * DK_; i += 256) st[i] = st[i] * cd + kvc[base + i];
        __syncthreads();
        if (tid < DK_) {
            float s = 0.f;
            for (int k2 = 0; k2 < DK_; ++k2) s += fabsf(st[k2 * DK_ + tid]);
            colsum[tid] = s;
        }
        __syncthreads();
        if (tid == 0) {
            float m = 1.f;
            for (int d2 = 0; d2 < DK_; ++d2) m = fmaxf(m, colsum[d2]);
            *scale_s = m;
        }
        __syncthreads();
    }
}

// ---------------- cross_out = (qr * inner_decay) @ kv_rec ----------------
__global__ void __launch_bounds__(256)
msr_cross(const float* __restrict__ qr, const float* __restrict__ kvrec,
          const float* __restrict__ inner_decay, float* __restrict__ cross_out) {
    int b = blockIdx.z, n = blockIdx.y, h = blockIdx.x;
    size_t rowbase = ((size_t)b * T_ + (size_t)n * CL_) * D_ + (size_t)h * DK_;
    const float* qb = qr + rowbase;
    const float* kv = kvrec + (((size_t)b * NC_ + n) * H_ + h) * (size_t)(DK_ * DK_);
    const float* idec = inner_decay + (size_t)h * CL_;
    int wave = threadIdx.x >> 5;
    int lane = threadIdx.x & 31, half = lane >> 4, col = lane & 15;
    for (int ls = 0; ls < 4; ++ls) {
        int l0 = (wave * 4 + ls) * 16;
        float dec = idec[l0 + (lane & 15)];
        v8f acc[8];
#pragma unroll
        for (int j = 0; j < 8; ++j) acc[j] = (v8f){};
#pragma unroll
        for (int k0 = 0; k0 < DK_; k0 += 32) {
            bf16x16 a = frag_f32(qb, D_, l0, k0, dec);
            bf16x16 bb[8];
#pragma unroll
            for (int j = 0; j < 8; ++j) bb[j] = frag_gather(kv, DK_, k0, j * 16, nullptr);
#pragma unroll
            for (int j = 0; j < 8; ++j) acc[j] = WMMA_BF16(a, bb[j], acc[j]);
        }
#pragma unroll
        for (int j = 0; j < 8; ++j)
#pragma unroll
            for (int i = 0; i < 8; ++i) {
                int l = l0 + i + 8 * half;
                cross_out[rowbase + (size_t)l * D_ + j * 16 + col] = acc[j][i];
            }
    }
}

// ---------------- combine + layernorm + silu gate -> bf16 ----------------
__global__ void msr_combine_ln_gate(const float* __restrict__ inner_out,
                                    const float* __restrict__ cross_out,
                                    const float* __restrict__ inner_scale,
                                    const float* __restrict__ cross_scale,
                                    const float* __restrict__ g, bf16_t* __restrict__ pre) {
    int wave = threadIdx.x >> 5, lane = threadIdx.x & 31;
    size_t row = (size_t)blockIdx.x * 8 + wave;          // (b*T+t)*H + h
    int h = (int)(row % H_);
    size_t bt = row / H_;
    int t = (int)(bt % T_);
    int b = (int)(bt / T_);
    int n = t / CL_, l = t % CL_;
    size_t base = ((size_t)b * T_ + t) * D_ + (size_t)h * DK_;
    float is = inner_scale[(((size_t)b * NC_ + n) * H_ + h) * CL_ + l];
    float cs = cross_scale[((size_t)b * NC_ + n) * H_ + h];
    float al = fmaxf(is, cs);
    float vals[4], sum = 0.f;
#pragma unroll
    for (int i = 0; i < 4; ++i) {
        int d = lane + i * 32;
        float o = inner_out[base + d] * (is / al) + cross_out[base + d] * (cs / al);
        vals[i] = o; sum += o;
    }
#pragma unroll
    for (int off = 16; off; off >>= 1) sum += __shfl_xor(sum, off, 32);
    float mu = sum / (float)DK_;
    float var = 0.f;
#pragma unroll
    for (int i = 0; i < 4; ++i) { float d = vals[i] - mu; var += d * d; }
#pragma unroll
    for (int off = 16; off; off >>= 1) var += __shfl_xor(var, off, 32);
    var /= (float)DK_;
    float inv = rsqrtf(var + 1e-6f);
#pragma unroll
    for (int i = 0; i < 4; ++i) {
        int d = lane + i * 32;
        float o = (vals[i] - mu) * inv;
        float gv = g[base + d];
        float sg = gv / (1.f + __expf(-gv));
        pre[base + d] = (__bf16)(sg * o);
    }
}

// ---------------- host-side launch ----------------
extern "C" void kernel_launch(void* const* d_in, const int* in_sizes, int n_in,
                              void* d_out, int out_size, void* d_ws, size_t ws_size,
                              hipStream_t stream) {
    const float* x   = (const float*)d_in[0];
    const float* sn  = (const float*)d_in[1];
    const float* cs  = (const float*)d_in[2];
    const float* mask        = (const float*)d_in[3];
    const float* cross_decay = (const float*)d_in[4];
    const float* inner_decay = (const float*)d_in[5];
    const float* wq = (const float*)d_in[6];  const float* bq = (const float*)d_in[7];
    const float* wk = (const float*)d_in[8];  const float* bk = (const float*)d_in[9];
    const float* wv = (const float*)d_in[10]; const float* bv = (const float*)d_in[11];
    const float* wg = (const float*)d_in[12]; const float* bg = (const float*)d_in[13];
    const float* wo = (const float*)d_in[14]; const float* bo = (const float*)d_in[15];
    float* out = (float*)d_out;

    const size_t NBT = (size_t)B_ * T_;
    const size_t NBTD = NBT * D_;
    const size_t NW = (size_t)D_ * D_;

    char* p = (char*)d_ws;
    auto carve = [&](size_t bytes) { void* r = (void*)p; p += (bytes + 255) & ~(size_t)255; return r; };
    bf16_t* xb  = (bf16_t*)carve(NBTD * 2);
    bf16_t* wqb = (bf16_t*)carve(NW * 2);
    bf16_t* wkb = (bf16_t*)carve(NW * 2);
    bf16_t* wvb = (bf16_t*)carve(NW * 2);
    bf16_t* wgb = (bf16_t*)carve(NW * 2);
    bf16_t* wob = (bf16_t*)carve(NW * 2);
    float* q = (float*)carve(NBTD * 4);
    float* k = (float*)carve(NBTD * 4);
    float* v = (float*)carve(NBTD * 4);
    float* gbuf = (float*)carve(NBTD * 4);
    float* inner_out = (float*)carve(NBTD * 4);
    float* cross_out = (float*)carve(NBTD * 4);
    float* inner_scale = (float*)carve((size_t)B_ * NC_ * H_ * CL_ * 4);
    float* kvc   = (float*)carve((size_t)B_ * NC_ * H_ * DK_ * DK_ * 4);
    float* kvrec = (float*)carve((size_t)B_ * NC_ * H_ * DK_ * DK_ * 4);
    float* cross_scale = (float*)carve((size_t)B_ * NC_ * H_ * 4);
    bf16_t* pre = (bf16_t*)carve(NBTD * 2);

    // 1. convert activations + weights to bf16
    msr_f2bf<<<dim3((unsigned)((NBTD + 255) / 256)), 256, 0, stream>>>(x, xb, NBTD);
    msr_f2bf<<<dim3((unsigned)((NW + 255) / 256)), 256, 0, stream>>>(wq, wqb, NW);
    msr_f2bf<<<dim3((unsigned)((NW + 255) / 256)), 256, 0, stream>>>(wk, wkb, NW);
    msr_f2bf<<<dim3((unsigned)((NW + 255) / 256)), 256, 0, stream>>>(wv, wvb, NW);
    msr_f2bf<<<dim3((unsigned)((NW + 255) / 256)), 256, 0, stream>>>(wg, wgb, NW);
    msr_f2bf<<<dim3((unsigned)((NW + 255) / 256)), 256, 0, stream>>>(wo, wob, NW);

    // 2. projections (async-pipelined WMMA GEMMs): M=8192, N=2048, K=2048
    dim3 ggrid(NBT / 128, D_ / 128);
    msr_gemm_bf16<<<ggrid, 256, 0, stream>>>(xb, wqb, bq, q,    (int)NBT, D_, D_, 1.f);
    msr_gemm_bf16<<<ggrid, 256, 0, stream>>>(xb, wkb, bk, k,    (int)NBT, D_, D_, SCALING_);
    msr_gemm_bf16<<<ggrid, 256, 0, stream>>>(xb, wvb, bv, v,    (int)NBT, D_, D_, 1.f);
    msr_gemm_bf16<<<ggrid, 256, 0, stream>>>(xb, wgb, bg, gbuf, (int)NBT, D_, D_, 1.f);

    // 3. theta shift q,k in place
    size_t npairs = NBTD / 2;
    msr_theta_shift<<<dim3((unsigned)((npairs + 255) / 256)), 256, 0, stream>>>(q, sn, cs);
    msr_theta_shift<<<dim3((unsigned)((npairs + 255) / 256)), 256, 0, stream>>>(k, sn, cs);

    // 4. per-chunk kv states, inner attention
    dim3 cgrid(H_, NC_, B_);
    msr_chunk_kv<<<cgrid, 256, 0, stream>>>(k, v, mask, kvc);
    msr_attn_inner<<<cgrid, 256, 0, stream>>>(q, k, v, mask, inner_out, inner_scale);

    // 5. sequential scan over chunks
    size_t scan_lds = (size_t)(DK_ * DK_ + DK_ + 1) * 4;
    msr_kv_scan<<<dim3(H_, B_), 256, scan_lds, stream>>>(kvc, cross_decay, kvrec, cross_scale);

    // 6. cross output
    msr_cross<<<cgrid, 256, 0, stream>>>(q, kvrec, inner_decay, cross_out);

    // 7. combine + layernorm + silu gate -> bf16 activations
    msr_combine_ln_gate<<<dim3((unsigned)(NBT * H_ / 8)), 256, 0, stream>>>(
        inner_out, cross_out, inner_scale, cross_scale, gbuf, pre);

    // 8. output projection: out = pre @ wo^T + bo
    msr_gemm_bf16<<<ggrid, 256, 0, stream>>>(pre, wob, bo, out, (int)NBT, D_, D_, 1.f);
}
